// LinearSAE_20340965114009
// MI455X (gfx1250) — compile-verified, matrix operand order checked by reading, not run on hardware
//
#include <hip/hip_runtime.h>

#define DM 2048        // d_model
#define DS 16384       // d_sparse
#define NROWS 16384    // N
#define KTOP 64u
#define LISTCAP 128

typedef __bf16 bf16_t;
typedef __attribute__((ext_vector_type(16))) __bf16 v16bf;
typedef __attribute__((ext_vector_type(8)))  __bf16 v8bf;
typedef __attribute__((ext_vector_type(8)))  float  v8f;

// ---------------------------------------------------------------------------
// Prep: hb = bf16(h - pre_bias), wb = bf16(W_enc)
// ---------------------------------------------------------------------------
__global__ __launch_bounds__(256) void sae_prep_h(const float* __restrict__ h,
                                                  const float* __restrict__ pre_bias,
                                                  bf16_t* __restrict__ hb) {
    int i = blockIdx.x * 256 + threadIdx.x;                 // total = NROWS*DM
    hb[i] = (bf16_t)(h[i] - pre_bias[i & (DM - 1)]);
}

__global__ __launch_bounds__(256) void sae_prep_w(const float* __restrict__ w,
                                                  bf16_t* __restrict__ wb) {
    int i = blockIdx.x * 256 + threadIdx.x;                 // total = DS*DM
    wb[i] = (bf16_t)w[i];
}

// ---------------------------------------------------------------------------
// Transpose W_dec [DM, DS] (row-major) -> W_decT bf16 [DS, DM]
// so decode gathers contiguous 4KB rows (L2-resident, 64MB < 192MB L2)
// ---------------------------------------------------------------------------
__global__ __launch_bounds__(256) void sae_transpose_wdec(const float* __restrict__ Wd,
                                                          bf16_t* __restrict__ WdT) {
    __shared__ float tile[32][33];
    const int sBase = blockIdx.x * 32;
    const int dBase = blockIdx.y * 32;
    const int tx = threadIdx.x & 31;
    const int ty = threadIdx.x >> 5;   // 0..7
#pragma unroll
    for (int i = 0; i < 4; ++i) {
        int d = dBase + ty + 8 * i;
        tile[ty + 8 * i][tx] = Wd[(size_t)d * DS + sBase + tx];
    }
    __syncthreads();
#pragma unroll
    for (int i = 0; i < 4; ++i) {
        int s = sBase + ty + 8 * i;
        WdT[(size_t)s * DM + dBase + tx] = (bf16_t)tile[tx][ty + 8 * i];
    }
}

// ---------------------------------------------------------------------------
// Encode GEMM: pre = (h-b) @ W_enc^T + enc_bias   via v_wmma_f32_16x16x32_bf16
// NT layout: A rows and B(=W_enc) rows are both K-contiguous.
// Wave tile 32x64 (2 A-frags x 4 B-frags, 8 wmma/K-step). WG = 8 waves = 64x256.
// ---------------------------------------------------------------------------
__device__ inline v16bf load_frag_bf16(const bf16_t* base, int k) {
    union { v16bf v; v8bf h[2]; } u;
    u.h[0] = *(const v8bf*)(base + k);        // K = k0..k0+7   (per-lane sub-block)
    u.h[1] = *(const v8bf*)(base + k + 16);   // K = k0+16..k0+23
    return u.v;
}

__global__ __launch_bounds__(256, 1) void sae_encode_wmma(
    const bf16_t* __restrict__ Ab,        // [NROWS, DM]
    const bf16_t* __restrict__ Bb,        // [DS, DM]
    const float* __restrict__ enc_bias,   // [DS]
    float* __restrict__ pre)              // [NROWS, DS] (acts region of d_out)
{
    const int lane = threadIdx.x & 31;
    const int wave = threadIdx.x >> 5;        // 0..7
    const int mw = wave & 1;                  // 2 M-waves
    const int nw = wave >> 1;                 // 4 N-waves
    const int mBase = blockIdx.x * 64 + mw * 32;
    const int nBase = blockIdx.y * 256 + nw * 64;

    const int lrow = lane & 15;
    const int lsub = (lane >> 4) * 8;         // interleaved K sub-block per ISA layout

    const bf16_t* aptr0 = Ab + (size_t)(mBase + lrow) * DM + lsub;
    const bf16_t* aptr1 = aptr0 + (size_t)16 * DM;
    const bf16_t* bptr  = Bb + (size_t)(nBase + lrow) * DM + lsub;

    v8f acc[2][4] = {};

#pragma unroll 2
    for (int k = 0; k < DM; k += 32) {
        v16bf a0 = load_frag_bf16(aptr0, k);
        v16bf a1 = load_frag_bf16(aptr1, k);
        v16bf b0 = load_frag_bf16(bptr,                     k);
        v16bf b1 = load_frag_bf16(bptr + (size_t)16 * DM,   k);
        v16bf b2 = load_frag_bf16(bptr + (size_t)32 * DM,   k);
        v16bf b3 = load_frag_bf16(bptr + (size_t)48 * DM,   k);

        acc[0][0] = __builtin_amdgcn_wmma_f32_16x16x32_bf16(false, a0, false, b0, (short)0, acc[0][0], false, false);
        acc[0][1] = __builtin_amdgcn_wmma_f32_16x16x32_bf16(false, a0, false, b1, (short)0, acc[0][1], false, false);
        acc[0][2] = __builtin_amdgcn_wmma_f32_16x16x32_bf16(false, a0, false, b2, (short)0, acc[0][2], false, false);
        acc[0][3] = __builtin_amdgcn_wmma_f32_16x16x32_bf16(false, a0, false, b3, (short)0, acc[0][3], false, false);
        acc[1][0] = __builtin_amdgcn_wmma_f32_16x16x32_bf16(false, a1, false, b0, (short)0, acc[1][0], false, false);
        acc[1][1] = __builtin_amdgcn_wmma_f32_16x16x32_bf16(false, a1, false, b1, (short)0, acc[1][1], false, false);
        acc[1][2] = __builtin_amdgcn_wmma_f32_16x16x32_bf16(false, a1, false, b2, (short)0, acc[1][2], false, false);
        acc[1][3] = __builtin_amdgcn_wmma_f32_16x16x32_bf16(false, a1, false, b3, (short)0, acc[1][3], false, false);
    }

    // Epilogue: C layout — VGPR r -> rows (r, r+8 by lane group), lane&15 -> column
#pragma unroll
    for (int i = 0; i < 2; ++i) {
#pragma unroll
        for (int j = 0; j < 4; ++j) {
            const int n = nBase + j * 16 + (lane & 15);
            const float bias = enc_bias[n];
            const int m0 = mBase + i * 16 + ((lane >> 4) << 3);
            float* outp = pre + (size_t)m0 * DS + n;
#pragma unroll
            for (int r = 0; r < 8; ++r)
                outp[(size_t)r * DS] = acc[i][j][r] + bias;
        }
    }
}

// ---------------------------------------------------------------------------
// Per-row exact top-64 threshold via 4-pass radix select on order-flipped
// float bits, then deterministic in-place ReLU scatter + compact (idx,val) list.
// One 256-thread block per row.
// ---------------------------------------------------------------------------
__device__ inline unsigned orderF(unsigned u) {
    unsigned mask = (unsigned)(((int)u) >> 31) | 0x80000000u;  // monotone map
    return u ^ mask;
}

__global__ __launch_bounds__(256) void sae_topk_select(
    float* __restrict__ pre,          // in: pre, out: acts (in place)
    int* __restrict__ listIdx,        // [NROWS, LISTCAP]
    float* __restrict__ listVal,      // [NROWS, LISTCAP]
    int* __restrict__ listCnt)        // [NROWS]
{
    __shared__ unsigned hist[256];
    __shared__ unsigned scan[257];
    __shared__ unsigned sPrefix, sRemain;

    const int row = blockIdx.x;
    const int t = threadIdx.x;
    float* rowp = pre + (size_t)row * DS;

    if (t == 0) { sPrefix = 0u; sRemain = KTOP; }

    for (int p = 3; p >= 0; --p) {
        hist[t] = 0u;
        __syncthreads();
        const unsigned prefix = sPrefix;
        for (int e = 0; e < DS / 256; ++e) {
            unsigned ou = orderF(__float_as_uint(rowp[t + 256 * e]));
            bool match = (p == 3) || ((ou >> (8 * (p + 1))) == (prefix >> (8 * (p + 1))));
            if (match) atomicAdd(&hist[(ou >> (8 * p)) & 255u], 1u);
        }
        __syncthreads();
        if (t == 0) {
            unsigned rem = sRemain, cum = 0u;
            int b = 255;
            for (; b > 0; --b) {
                if (cum + hist[b] >= rem) break;
                cum += hist[b];
            }
            sPrefix = sPrefix | ((unsigned)b << (8 * p));
            sRemain = rem - cum;
        }
        __syncthreads();
    }
    const unsigned thresh = sPrefix;

    // Phase 1: deterministic per-thread counts of positive selected elements
    unsigned cnt = 0;
    for (int e = 0; e < DS / 256; ++e) {
        float v = rowp[t + 256 * e];
        unsigned ou = orderF(__float_as_uint(v));
        if (ou >= thresh && v > 0.0f) cnt++;
    }
    scan[t] = cnt;
    __syncthreads();
    if (t == 0) {
        unsigned s = 0;
        for (int i = 0; i < 256; ++i) { unsigned c = scan[i]; scan[i] = s; s += c; }
        scan[256] = s;
        listCnt[row] = (int)(s < LISTCAP ? s : LISTCAP);
    }
    __syncthreads();

    // Phase 2: in-place ReLU scatter + list write at deterministic positions
    unsigned pos = scan[t];
    for (int e = 0; e < DS / 256; ++e) {
        const int sidx = t + 256 * e;
        float v = rowp[sidx];
        unsigned ou = orderF(__float_as_uint(v));
        bool keep = (ou >= thresh) && (v > 0.0f);
        rowp[sidx] = keep ? v : 0.0f;
        if (keep) {
            if (pos < LISTCAP) {
                listIdx[(size_t)row * LISTCAP + pos] = sidx;
                listVal[(size_t)row * LISTCAP + pos] = v;
            }
            pos++;
        }
    }
}

// ---------------------------------------------------------------------------
// Sparse decode: recon[n,:] = pre_bias + sum_j val_j * W_decT[idx_j, :]
// One block per row; thread owns 8 dims (stride-256 coalesced); W_decT in L2.
// ---------------------------------------------------------------------------
__global__ __launch_bounds__(256) void sae_decode(
    const bf16_t* __restrict__ WdT,       // [DS, DM]
    const float* __restrict__ pre_bias,   // [DM]
    const int* __restrict__ listIdx,
    const float* __restrict__ listVal,
    const int* __restrict__ listCnt,
    float* __restrict__ recon)            // [NROWS, DM]
{
    __shared__ int   sIdx[LISTCAP];
    __shared__ float sVal[LISTCAP];
    const int row = blockIdx.x;
    const int t = threadIdx.x;
    const int cnt = listCnt[row];
    if (t < LISTCAP) {
        sIdx[t] = (t < cnt) ? listIdx[(size_t)row * LISTCAP + t] : 0;
        sVal[t] = (t < cnt) ? listVal[(size_t)row * LISTCAP + t] : 0.0f;
    }
    __syncthreads();

    float acc[DM / 256];
#pragma unroll
    for (int u = 0; u < DM / 256; ++u) acc[u] = pre_bias[t + 256 * u];

    for (int j = 0; j < cnt; ++j) {
        const bf16_t* wrow = WdT + (size_t)sIdx[j] * DM;
        const float v = sVal[j];
#pragma unroll
        for (int u = 0; u < DM / 256; ++u)
            acc[u] += v * (float)wrow[t + 256 * u];
    }

    float* out = recon + (size_t)row * DM;
#pragma unroll
    for (int u = 0; u < DM / 256; ++u) out[t + 256 * u] = acc[u];
}

// ---------------------------------------------------------------------------
// Launcher
// ---------------------------------------------------------------------------
extern "C" void kernel_launch(void* const* d_in, const int* in_sizes, int n_in,
                              void* d_out, int out_size, void* d_ws, size_t ws_size,
                              hipStream_t stream) {
    const float* h        = (const float*)d_in[0];  // [NROWS, DM]
    const float* W_enc    = (const float*)d_in[1];  // [DS, DM]
    const float* W_dec    = (const float*)d_in[2];  // [DM, DS]
    const float* pre_bias = (const float*)d_in[3];  // [DM]
    const float* enc_bias = (const float*)d_in[4];  // [DS]

    float* acts  = (float*)d_out;                       // [NROWS, DS]
    float* recon = acts + (size_t)NROWS * DS;           // [NROWS, DM]

    char* ws = (char*)d_ws;
    const size_t szHB  = (size_t)NROWS * DM * sizeof(bf16_t);   // 64 MB
    const size_t szWB  = (size_t)DS * DM * sizeof(bf16_t);      // 64 MB
    const size_t szWDT = (size_t)DS * DM * sizeof(bf16_t);      // 64 MB
    bf16_t* hb  = (bf16_t*)(ws);
    bf16_t* wb  = (bf16_t*)(ws + szHB);
    bf16_t* wdT = (bf16_t*)(ws + szHB + szWB);
    char* p = ws + szHB + szWB + szWDT;
    int*   listIdx = (int*)p;                 p += (size_t)NROWS * LISTCAP * sizeof(int);
    float* listVal = (float*)p;               p += (size_t)NROWS * LISTCAP * sizeof(float);
    int*   listCnt = (int*)p;

    // 1) prep bf16 operands
    sae_prep_h<<<(NROWS * DM) / 256, 256, 0, stream>>>(h, pre_bias, hb);
    sae_prep_w<<<(DS * DM) / 256, 256, 0, stream>>>(W_enc, wb);

    // 2) transpose W_dec -> bf16 [DS, DM]
    sae_transpose_wdec<<<dim3(DS / 32, DM / 32), 256, 0, stream>>>(W_dec, wdT);

    // 3) encode GEMM (WMMA bf16) -> pre written into acts region
    sae_encode_wmma<<<dim3(NROWS / 64, DS / 256), 256, 0, stream>>>(hb, wb, enc_bias, acts);

    // 4) per-row top-64 select: in-place ReLU scatter + compact lists
    sae_topk_select<<<NROWS, 256, 0, stream>>>(acts, listIdx, listVal, listCnt);

    // 5) sparse decode
    sae_decode<<<NROWS, 256, 0, stream>>>(wdT, pre_bias, listIdx, listVal, listCnt, recon);
}
